// PointTransformerLayer_17265768530257
// MI455X (gfx1250) — matrix operand order repaired and checked by reading, not compiled, not run on hardware
//
#include <hip/hip_runtime.h>
#include <stdint.h>

// ---------------------------------------------------------------------------
// Point Transformer layer, fused for MI455X (gfx1250, wave32, WMMA).
// GEMMs in bf16 (fp32 accumulate) via v_wmma_f32_16x16x32_bf16.
// One wave == one point: its K=16 neighbors form one 16-row WMMA tile.
// ---------------------------------------------------------------------------

#define CDIM 128
#define KNB  16
#define TS   136   // bf16 LDS tile row stride (elements, even -> u32-aligned)
#define PS   132   // f32 pos_enc LDS row stride

typedef __attribute__((ext_vector_type(16))) __bf16 v16bf;
typedef __attribute__((ext_vector_type(8)))  float  v8f;

union ABFrag { v16bf v; uint32_t u[8]; };

__device__ __forceinline__ uint16_t f2bf(float f) {
  union { float f; uint32_t u; } x; x.f = f;
  uint32_t u = x.u + 0x7FFFu + ((x.u >> 16) & 1u);   // round-to-nearest-even
  return (uint16_t)(u >> 16);
}

__device__ __forceinline__ v8f wmma_bf16(v16bf a, v16bf b, v8f c) {
  // D = A(16x32) * B(32x16) + C(16x16 f32)
  return __builtin_amdgcn_wmma_f32_16x16x32_bf16(
      /*neg_a=*/false, a, /*neg_b=*/false, b,
      /*c_mod=*/(short)0, c, /*reuse_a=*/false, /*reuse_b=*/false);
}

// A-fragment, contiguous K order (matches load_b_frag):
//   lanes 0-15: M=lane, V0..V7 = K 0..15 (+16 for lanes 16-31)
__device__ __forceinline__ v16bf load_a_frag(const uint16_t* tile, int kc, int lane) {
  const int hi = (lane >> 4) & 1;
  const int m  = lane & 15;
  const uint16_t* r = tile + m * TS + kc * 32 + hi * 16;
  ABFrag a;
#pragma unroll
  for (int i = 0; i < 4; ++i) a.u[i] = *(const uint32_t*)(r + 2 * i);
#pragma unroll
  for (int i = 0; i < 4; ++i) a.u[4 + i] = *(const uint32_t*)(r + 8 + 2 * i);
  return a.v;
}

// B-fragment (32x16, K x N). Weights stored row-major W[n][k] (128x128 bf16);
// B[k][n] = W[n][k] -> lane (N = nb*16 + lane%16) reads contiguous K pairs.
__device__ __forceinline__ v16bf load_b_frag(const uint16_t* w, int nb, int kc, int lane) {
  const int hi = (lane >> 4) & 1;
  const int n  = nb * 16 + (lane & 15);
  const uint16_t* r = w + n * CDIM + kc * 32 + hi * 16;
  ABFrag b;
#pragma unroll
  for (int i = 0; i < 8; ++i) b.u[i] = *(const uint32_t*)(r + 2 * i);
  return b.v;
}

// ---------------------------------------------------------------------------
__global__ void cvt_bf16_kernel(const float* __restrict__ src,
                                uint16_t* __restrict__ dst, int n) {
  int i = blockIdx.x * blockDim.x + threadIdx.x;
  if (i < n) dst[i] = f2bf(src[i]);
}

// ---------------------------------------------------------------------------
// Generic  out[M x 128] = in[M x 128] @ W^T + bias   (bf16 WMMA, fp32 acc)
// 128 threads = 4 waves; one 16-row tile per wave per iteration.
__global__ void __launch_bounds__(128)
proj_gemm_kernel(const float* __restrict__ in, const uint16_t* __restrict__ w16,
                 const float* __restrict__ bias, float* __restrict__ out, int M) {
  __shared__ uint16_t sW[CDIM * CDIM];
  __shared__ float    sB[CDIM];
  __shared__ uint16_t sA[4][16 * TS];
  const int t = threadIdx.x;
  const int wave = t >> 5, lane = t & 31;

  for (int i = t; i < (CDIM * CDIM) / 2; i += blockDim.x)
    ((uint32_t*)sW)[i] = ((const uint32_t*)w16)[i];
  if (t < CDIM) sB[t] = bias[t];
  __syncthreads();

  const int hi = lane >> 4, nlo = lane & 15;
  const int nTiles = M >> 4;
  const int waveStride = gridDim.x * 4;
  for (int tile = blockIdx.x * 4 + wave; tile < nTiles; tile += waveStride) {
    const float* src = in + (size_t)tile * (16 * CDIM);
    const int nxt = tile + waveStride;
    if (nxt < nTiles)  // stream next A-tile into cache (global_prefetch_b8)
      __builtin_prefetch(in + (size_t)nxt * (16 * CDIM) + lane * 64, 0, 0);

    uint16_t* tA = sA[wave];
    for (int i = lane; i < 16 * CDIM; i += 32) {
      const int r = i >> 7, c = i & (CDIM - 1);
      tA[r * TS + c] = f2bf(src[i]);
    }
    asm volatile("s_wait_dscnt 0" ::: "memory");   // LDS tile visible wave-wide

    v16bf a0 = load_a_frag(tA, 0, lane), a1 = load_a_frag(tA, 1, lane);
    v16bf a2 = load_a_frag(tA, 2, lane), a3 = load_a_frag(tA, 3, lane);

    float* obase = out + (size_t)tile * (16 * CDIM);
#pragma unroll
    for (int nb = 0; nb < 8; ++nb) {
      const float bv = sB[nb * 16 + nlo];
      v8f acc;
#pragma unroll
      for (int r = 0; r < 8; ++r) acc[r] = bv;
      acc = wmma_bf16(a0, load_b_frag(sW, nb, 0, lane), acc);
      acc = wmma_bf16(a1, load_b_frag(sW, nb, 1, lane), acc);
      acc = wmma_bf16(a2, load_b_frag(sW, nb, 2, lane), acc);
      acc = wmma_bf16(a3, load_b_frag(sW, nb, 3, lane), acc);
      float* op = obase + nb * 16 + nlo;
#pragma unroll
      for (int r = 0; r < 8; ++r) op[(size_t)(r + 8 * hi) * CDIM] = acc[r];
    }
  }
}

// ---------------------------------------------------------------------------
#define SM_FIXED_BYTES (3 * 32768 + (384 + 4 * 128) * 4)        /* 101888 */
#define PERWAVE_BYTES  (16 * TS * 2 + 16 * PS * 4 + 64)         /* 12864  */
#define SMEM_TOTAL     (SM_FIXED_BYTES + 4 * PERWAVE_BYTES)     /* 153344 */

__global__ void __launch_bounds__(128)
ptrans_attn_kernel(const float* __restrict__ xyz, const int* __restrict__ nbr,
                   const float* __restrict__ q, const float* __restrict__ k,
                   const float* __restrict__ v,
                   const uint16_t* __restrict__ wd2, const uint16_t* __restrict__ wg1,
                   const uint16_t* __restrict__ wg2,
                   const float* __restrict__ Wd1, const float* __restrict__ bd1,
                   const float* __restrict__ bd2, const float* __restrict__ bg1,
                   const float* __restrict__ bg2,
                   float* __restrict__ outmid, int BN, int Npts) {
  extern __shared__ uint8_t smem[];
  uint16_t* sWd2 = (uint16_t*)smem;
  uint16_t* sWg1 = sWd2 + CDIM * CDIM;
  uint16_t* sWg2 = sWg1 + CDIM * CDIM;
  float* sWd1 = (float*)(sWg2 + CDIM * CDIM);   // 128x3 fp32
  float* sbd1 = sWd1 + 384;
  float* sbd2 = sbd1 + CDIM;
  float* sbg1 = sbd2 + CDIM;
  float* sbg2 = sbg1 + CDIM;

  const int t = threadIdx.x;
  const int wave = t >> 5, lane = t & 31;
  uint8_t* pw   = smem + SM_FIXED_BYTES + wave * PERWAVE_BYTES;
  uint16_t* tA  = (uint16_t*)pw;                           // 16 x 128 bf16 tile
  float*    posb = (float*)(pw + 16 * TS * 2);             // 16 x 128 f32 pos_enc
  int*      idxb = (int*)(pw + 16 * TS * 2 + 16 * PS * 4); // 16 neighbor ids

  for (int i = t; i < (CDIM * CDIM) / 2; i += blockDim.x) {
    ((uint32_t*)sWd2)[i] = ((const uint32_t*)wd2)[i];
    ((uint32_t*)sWg1)[i] = ((const uint32_t*)wg1)[i];
    ((uint32_t*)sWg2)[i] = ((const uint32_t*)wg2)[i];
  }
  for (int i = t; i < 384; i += blockDim.x) sWd1[i] = Wd1[i];
  if (t < CDIM) { sbd1[t] = bd1[t]; sbd2[t] = bd2[t]; sbg1[t] = bg1[t]; sbg2[t] = bg2[t]; }
  __syncthreads();

  const int hi = lane >> 4, nlo = lane & 15;
  const int wavesTotal = gridDim.x * 4;
  for (int p = blockIdx.x * 4 + wave; p < BN; p += wavesTotal) {
    const int b = p / Npts;
    const int bBase = b * Npts;
    const size_t rowOff = (size_t)p * CDIM;

    if (lane < KNB) idxb[lane] = nbr[(size_t)p * KNB + lane];
    asm volatile("s_wait_dscnt 0" ::: "memory");
    int nrow[8];                       // rows this lane needs (M = r + 8*hi)
#pragma unroll
    for (int r = 0; r < 8; ++r) nrow[r] = bBase + idxb[r + 8 * hi];

    // ---- pos-enc layer 1: Linear(3->128) + ReLU (VALU), tile -> LDS bf16
    const float cx = xyz[(size_t)p * 3 + 0], cy = xyz[(size_t)p * 3 + 1],
                cz = xyz[(size_t)p * 3 + 2];
    for (int m = 0; m < KNB; ++m) {
      const int g = bBase + idxb[m];
      const float d0 = cx - xyz[(size_t)g * 3 + 0];
      const float d1 = cy - xyz[(size_t)g * 3 + 1];
      const float d2 = cz - xyz[(size_t)g * 3 + 2];
#pragma unroll
      for (int cc = 0; cc < 4; ++cc) {
        const int c = lane + 32 * cc;
        const float* wr = sWd1 + c * 3;
        float h = sbd1[c] + d0 * wr[0] + d1 * wr[1] + d2 * wr[2];
        tA[m * TS + c] = f2bf(h > 0.f ? h : 0.f);
      }
    }
    asm volatile("s_wait_dscnt 0" ::: "memory");

    // ---- pos-enc layer 2 (WMMA) ; fuse build of attn_feat = q - k_n + pos
    v16bf a0 = load_a_frag(tA, 0, lane), a1 = load_a_frag(tA, 1, lane);
    v16bf a2 = load_a_frag(tA, 2, lane), a3 = load_a_frag(tA, 3, lane);
    const float* qrow = q + rowOff;
#pragma unroll
    for (int nb = 0; nb < 8; ++nb) {
      const int c = nb * 16 + nlo;
      v8f acc; { const float b2 = sbd2[c];
#pragma unroll
        for (int r = 0; r < 8; ++r) acc[r] = b2; }
      acc = wmma_bf16(a0, load_b_frag(sWd2, nb, 0, lane), acc);
      acc = wmma_bf16(a1, load_b_frag(sWd2, nb, 1, lane), acc);
      acc = wmma_bf16(a2, load_b_frag(sWd2, nb, 2, lane), acc);
      acc = wmma_bf16(a3, load_b_frag(sWd2, nb, 3, lane), acc);
      const float qv = qrow[c];
#pragma unroll
      for (int r = 0; r < 8; ++r) {
        const int m = r + 8 * hi;
        const float pos = acc[r];
        posb[m * PS + c] = pos;                               // keep fp32
        const float kv = k[(size_t)nrow[r] * CDIM + c];       // L2-resident gather
        tA[m * TS + c] = f2bf(qv - kv + pos);
      }
    }
    asm volatile("s_wait_dscnt 0" ::: "memory");

    // ---- gamma layer 1 + ReLU
    a0 = load_a_frag(tA, 0, lane); a1 = load_a_frag(tA, 1, lane);
    a2 = load_a_frag(tA, 2, lane); a3 = load_a_frag(tA, 3, lane);
#pragma unroll
    for (int nb = 0; nb < 8; ++nb) {
      const int c = nb * 16 + nlo;
      v8f acc; { const float b2 = sbg1[c];
#pragma unroll
        for (int r = 0; r < 8; ++r) acc[r] = b2; }
      acc = wmma_bf16(a0, load_b_frag(sWg1, nb, 0, lane), acc);
      acc = wmma_bf16(a1, load_b_frag(sWg1, nb, 1, lane), acc);
      acc = wmma_bf16(a2, load_b_frag(sWg1, nb, 2, lane), acc);
      acc = wmma_bf16(a3, load_b_frag(sWg1, nb, 3, lane), acc);
#pragma unroll
      for (int r = 0; r < 8; ++r)
        tA[(r + 8 * hi) * TS + c] = f2bf(acc[r] > 0.f ? acc[r] : 0.f);
    }
    asm volatile("s_wait_dscnt 0" ::: "memory");

    // ---- gamma layer 2 + softmax over 16 neighbors (M-dim) + weighted sum
    a0 = load_a_frag(tA, 0, lane); a1 = load_a_frag(tA, 1, lane);
    a2 = load_a_frag(tA, 2, lane); a3 = load_a_frag(tA, 3, lane);
    float outv[8];
#pragma unroll
    for (int nb = 0; nb < 8; ++nb) {
      const int c = nb * 16 + nlo;
      v8f acc; { const float b2 = sbg2[c];
#pragma unroll
        for (int r = 0; r < 8; ++r) acc[r] = b2; }
      acc = wmma_bf16(a0, load_b_frag(sWg2, nb, 0, lane), acc);
      acc = wmma_bf16(a1, load_b_frag(sWg2, nb, 1, lane), acc);
      acc = wmma_bf16(a2, load_b_frag(sWg2, nb, 2, lane), acc);
      acc = wmma_bf16(a3, load_b_frag(sWg2, nb, 3, lane), acc);
      // softmax across M: 8 values in this lane + 8 in partner (lane ^ 16)
      float mx = acc[0];
#pragma unroll
      for (int r = 1; r < 8; ++r) mx = fmaxf(mx, acc[r]);
      mx = fmaxf(mx, __shfl_xor(mx, 16, 32));
      float e[8], s = 0.f;
#pragma unroll
      for (int r = 0; r < 8; ++r) { e[r] = __expf(acc[r] - mx); s += e[r]; }
      s += __shfl_xor(s, 16, 32);
      const float inv = 1.0f / s;
      float part = 0.f;
#pragma unroll
      for (int r = 0; r < 8; ++r) {
        const int m = r + 8 * hi;
        const float vv = v[(size_t)nrow[r] * CDIM + c] + posb[m * PS + c];
        part += e[r] * vv;
      }
      part *= inv;
      part += __shfl_xor(part, 16, 32);   // add partner half of the M-sum
      outv[nb] = part;
    }
    if (hi == 0) {
      float* dst = outmid + rowOff;
#pragma unroll
      for (int nb = 0; nb < 8; ++nb) dst[nb * 16 + nlo] = outv[nb];
    }
  }
}

// ---------------------------------------------------------------------------
extern "C" void kernel_launch(void* const* d_in, const int* in_sizes, int n_in,
                              void* d_out, int out_size, void* d_ws, size_t ws_size,
                              hipStream_t stream) {
  const float* xyz  = (const float*)d_in[0];
  const float* feat = (const float*)d_in[1];
  const int*   nbr  = (const int*)d_in[2];
  const float* Wq  = (const float*)d_in[3];  const float* bq  = (const float*)d_in[4];
  const float* Wk  = (const float*)d_in[5];  const float* bk  = (const float*)d_in[6];
  const float* Wv  = (const float*)d_in[7];  const float* bv  = (const float*)d_in[8];
  const float* Wd1 = (const float*)d_in[9];  const float* bd1 = (const float*)d_in[10];
  const float* Wd2 = (const float*)d_in[11]; const float* bd2 = (const float*)d_in[12];
  const float* Wg1 = (const float*)d_in[13]; const float* bg1 = (const float*)d_in[14];
  const float* Wg2 = (const float*)d_in[15]; const float* bg2 = (const float*)d_in[16];
  const float* Wo  = (const float*)d_in[17]; const float* bo  = (const float*)d_in[18];

  const int BN   = in_sizes[1] / CDIM;                 // B*N = 32768
  const int Npts = (BN % 2 == 0) ? (BN / 2) : BN;      // reference B == 2

  // workspace layout: 7 bf16 weight mats (32 KB each) + q,k,v,mid fp32 (~17 MB each)
  uint8_t* ws = (uint8_t*)d_ws;
  const size_t WB = (size_t)CDIM * CDIM * sizeof(uint16_t);
  uint16_t* wq16 = (uint16_t*)(ws + 0 * WB);
  uint16_t* wk16 = (uint16_t*)(ws + 1 * WB);
  uint16_t* wv16 = (uint16_t*)(ws + 2 * WB);
  uint16_t* wd216 = (uint16_t*)(ws + 3 * WB);
  uint16_t* wg116 = (uint16_t*)(ws + 4 * WB);
  uint16_t* wg216 = (uint16_t*)(ws + 5 * WB);
  uint16_t* wo16  = (uint16_t*)(ws + 6 * WB);
  const size_t FB = (size_t)BN * CDIM * sizeof(float);
  float* qbuf   = (float*)(ws + 7 * WB);
  float* kbuf   = (float*)(ws + 7 * WB + 1 * FB);
  float* vbuf   = (float*)(ws + 7 * WB + 2 * FB);
  float* midbuf = (float*)(ws + 7 * WB + 3 * FB);
  (void)ws_size; (void)n_in; (void)out_size;

  const int nW = CDIM * CDIM;
  cvt_bf16_kernel<<<(nW + 255) / 256, 256, 0, stream>>>(Wq,  wq16,  nW);
  cvt_bf16_kernel<<<(nW + 255) / 256, 256, 0, stream>>>(Wk,  wk16,  nW);
  cvt_bf16_kernel<<<(nW + 255) / 256, 256, 0, stream>>>(Wv,  wv16,  nW);
  cvt_bf16_kernel<<<(nW + 255) / 256, 256, 0, stream>>>(Wd2, wd216, nW);
  cvt_bf16_kernel<<<(nW + 255) / 256, 256, 0, stream>>>(Wg1, wg116, nW);
  cvt_bf16_kernel<<<(nW + 255) / 256, 256, 0, stream>>>(Wg2, wg216, nW);
  cvt_bf16_kernel<<<(nW + 255) / 256, 256, 0, stream>>>(Wo,  wo16,  nW);

  // q / k / v projections
  proj_gemm_kernel<<<512, 128, 0, stream>>>(feat, wq16, bq, qbuf, BN);
  proj_gemm_kernel<<<512, 128, 0, stream>>>(feat, wk16, bk, kbuf, BN);
  proj_gemm_kernel<<<512, 128, 0, stream>>>(feat, wv16, bv, vbuf, BN);

  // fused pos-enc + vector attention (150 KB dynamic LDS < 320 KB/WGP)
  (void)hipFuncSetAttribute(reinterpret_cast<const void*>(ptrans_attn_kernel),
                            hipFuncAttributeMaxDynamicSharedMemorySize, SMEM_TOTAL);
  ptrans_attn_kernel<<<2048, 128, SMEM_TOTAL, stream>>>(
      xyz, nbr, qbuf, kbuf, vbuf, wd216, wg116, wg216,
      Wd1, bd1, bd2, bg1, bg2, midbuf, BN, Npts);

  // fc_out
  proj_gemm_kernel<<<512, 128, 0, stream>>>(midbuf, wo16, bo, (float*)d_out, BN);
}
